// EncoderSimpleRNN_69733089018329
// MI455X (gfx1250) — compile-verified, matrix operand order checked by reading, not run on hardware
//
#include <hip/hip_runtime.h>
#include <hip/hip_bf16.h>
#include <stdint.h>

// ---------------------------------------------------------------------------
// EncoderSimpleRNN for MI455X (gfx1250, wave32, WMMA bf16, async-LDS)
//   B=256, T=128, H=512, VOCAB=32000
//   Phase 0: convert W_ih, W_hh, h0 to bf16 (workspace)
//   Phase 1: x_proj = gather(emb, input) @ W_ih^T + b_ih   -> written to d_out
//            (4 N-tiles per wave, software-pipelined fragment loads)
//   Phase 2: 128 sequential steps: h = tanh(x_t + h @ W_hh^T + b_hh)
//            h-strip staged with GLOBAL_LOAD_ASYNC_TO_LDS_B128; pipelined
// ---------------------------------------------------------------------------

typedef __attribute__((ext_vector_type(16))) __bf16 v16bf;
typedef __attribute__((ext_vector_type(8)))  float  v8f;

#define HID   512
#define BATCH 256
#define SEQ   128
#define MROWS (BATCH * SEQ)   // 32768

// One 16x32 bf16 fragment = 8 dwords per lane = two 16-byte chunks:
//   dwords [K0/2 + 4*half .. +3] and [K0/2 + 8 + 4*half .. +3]
union FragB16 { v16bf v; uint4 q[2]; uint32_t u[8]; };

// ---------------------------------------------------------------------------
// Phase 0: fp32 -> bf16 conversions (weights once, initial hidden once)
// ---------------------------------------------------------------------------
__global__ __launch_bounds__(256) void prep_kernel(
    const float* __restrict__ Wih, const float* __restrict__ Whh,
    const float* __restrict__ h0,
    __bf16* __restrict__ Wih_b, __bf16* __restrict__ Whh_b,
    __bf16* __restrict__ h_b)
{
    int idx = blockIdx.x * 256 + threadIdx.x;
    if (idx < HID * HID) {
        Wih_b[idx] = (__bf16)Wih[idx];
        Whh_b[idx] = (__bf16)Whh[idx];
    }
    if (idx < BATCH * HID) {
        h_b[idx] = (__bf16)h0[idx];
    }
}

// ---------------------------------------------------------------------------
// Phase 1: x_proj[m, n] = sum_k emb[input[m], k] * W_ih[n, k] + b_ih[n]
//   grid = 2048 blocks (one 16-row strip each); block = 256 thr = 8 wave32s
//   Each wave computes 4 adjacent 16x16 tiles (16x64); 8 waves cover N=512.
//   A strip (16 x 512, gathered + cvt to bf16) staged in LDS, shared by waves.
//   K-loop is double-buffered: loads for kk+1 issue before WMMAs of kk.
// ---------------------------------------------------------------------------
__global__ __launch_bounds__(256) void xproj_kernel(
    const int*    __restrict__ input,
    const float*  __restrict__ emb,
    const __bf16* __restrict__ Wb,
    const float*  __restrict__ bias,
    float*        __restrict__ out)
{
    __shared__ __bf16 As[16 * HID];     // 16 KB

    const int tid = threadIdx.x;
    const long m0 = (long)blockIdx.x * 16;

    // --- stage gathered A strip: 16 rows x 512 cols, fp32 -> bf16 ---
    {
        const int row  = tid >> 4;            // 0..15
        const int cseg = (tid & 15) * 32;     // 32 cols per thread
        const long erow = (long)input[m0 + row];
        const float* src = emb + erow * HID + cseg;
        __bf16* dst = As + row * HID + cseg;
#pragma unroll
        for (int j = 0; j < 32; j += 4) {
            float4 f = *(const float4*)(src + j);
            dst[j + 0] = (__bf16)f.x;
            dst[j + 1] = (__bf16)f.y;
            dst[j + 2] = (__bf16)f.z;
            dst[j + 3] = (__bf16)f.w;
        }
    }
    __syncthreads();

    const int wave = tid >> 5;
    const int lane = tid & 31;
    const int half = lane >> 4;
    const int mn   = lane & 15;           // A-row for frag A, B-col for frag B
    const int nb   = wave * 64;           // this wave's 64-column group

    const uint4* Aq = (const uint4*)(As + mn * HID);
    const uint4* Wq[4] = {
        (const uint4*)(Wb + (long)(nb +  0 + mn) * HID),
        (const uint4*)(Wb + (long)(nb + 16 + mn) * HID),
        (const uint4*)(Wb + (long)(nb + 32 + mn) * HID),
        (const uint4*)(Wb + (long)(nb + 48 + mn) * HID)
    };

    FragB16 a[2], b[2][4];
    // prologue: load fragments for kk = 0
    {
        const int i0 = half;
        a[0].q[0] = Aq[i0];
        a[0].q[1] = Aq[i0 + 2];
#pragma unroll
        for (int j = 0; j < 4; ++j) {
            b[0][j].q[0] = Wq[j][i0];
            b[0][j].q[1] = Wq[j][i0 + 2];
        }
    }

    v8f c0 = {}, c1 = {}, c2 = {}, c3 = {};
#pragma unroll
    for (int kk = 0; kk < 16; ++kk) {
        const int cur = kk & 1;
        const int nxt = cur ^ 1;
        if (kk < 15) {
            const int i1 = 4 * (kk + 1) + half;   // next chunk index
            a[nxt].q[0] = Aq[i1];
            a[nxt].q[1] = Aq[i1 + 2];
#pragma unroll
            for (int j = 0; j < 4; ++j) {
                b[nxt][j].q[0] = Wq[j][i1];
                b[nxt][j].q[1] = Wq[j][i1 + 2];
            }
        }
        c0 = __builtin_amdgcn_wmma_f32_16x16x32_bf16(false, a[cur].v, false, b[cur][0].v, (short)0, c0, false, false);
        c1 = __builtin_amdgcn_wmma_f32_16x16x32_bf16(false, a[cur].v, false, b[cur][1].v, (short)0, c1, false, false);
        c2 = __builtin_amdgcn_wmma_f32_16x16x32_bf16(false, a[cur].v, false, b[cur][2].v, (short)0, c2, false, false);
        c3 = __builtin_amdgcn_wmma_f32_16x16x32_bf16(false, a[cur].v, false, b[cur][3].v, (short)0, c3, false, false);
    }

    // Store 4 tiles (+ bias)
    v8f cc[4] = {c0, c1, c2, c3};
#pragma unroll
    for (int j = 0; j < 4; ++j) {
        const int col = nb + j * 16 + mn;
        const float bi = bias[col];
        float* outp = out + m0 * HID + col;
#pragma unroll
        for (int r = 0; r < 8; ++r)
            outp[(long)(r + 8 * half) * HID] = cc[j][r] + bi;
    }
}

// ---------------------------------------------------------------------------
// Phase 2 (one timestep): h_new = tanh(x_t + h_prev @ W_hh^T + b_hh)
//   xbase points at d_out + t*HID ; row stride in x/out is SEQ*HID.
//   grid = (16, 4) ; block = 256 = 8 waves, one 16x16 tile per wave.
//   h_prev strip (16 x 512 bf16 = 16 KB contiguous) staged with
//   GLOBAL_LOAD_ASYNC_TO_LDS_B128 (ASYNCcnt), then s_wait_asynccnt 0.
//   K-loop double-buffered like phase 1.
// ---------------------------------------------------------------------------
__global__ __launch_bounds__(256) void rnn_step_kernel(
    const __bf16* __restrict__ hprev_b,
    const __bf16* __restrict__ Whh_b,
    const float*  __restrict__ bhh,
    float*        __restrict__ xbase,      // d_out + t*HID (read-modify-write)
    __bf16*       __restrict__ hnext_b,
    float*        __restrict__ hfinal,
    int           writeFinal)
{
    __shared__ __bf16 Hs[16 * HID];     // 16 KB

    const int tid = threadIdx.x;
    const int m0  = blockIdx.x * 16;

    // --- async stage h_prev strip: 16 KB contiguous, 4 x b128 per thread ---
    {
        const unsigned lds_base = (unsigned)(uintptr_t)(void*)Hs;   // LDS offset
        const char* src = (const char*)(hprev_b + (long)m0 * HID);
#pragma unroll
        for (int p = 0; p < 4; ++p) {
            const unsigned boff = (unsigned)(tid + p * 256) * 16u;
            unsigned long long ga = (unsigned long long)(uintptr_t)(src + boff);
            unsigned la = lds_base + boff;
            asm volatile("global_load_async_to_lds_b128 %0, %1, off"
                         :: "v"(la), "v"(ga) : "memory");
        }
        asm volatile("s_wait_asynccnt 0" ::: "memory");
    }
    __syncthreads();

    const int wave = tid >> 5;
    const int lane = tid & 31;
    const int half = lane >> 4;
    const int mn   = lane & 15;
    const int n0   = blockIdx.y * 128 + wave * 16;

    const uint4* Hq = (const uint4*)(Hs + mn * HID);
    const uint4* Wq = (const uint4*)(Whh_b + (long)(n0 + mn) * HID);

    FragB16 a[2], b[2];
    {
        const int i0 = half;
        a[0].q[0] = Hq[i0]; a[0].q[1] = Hq[i0 + 2];
        b[0].q[0] = Wq[i0]; b[0].q[1] = Wq[i0 + 2];
    }

    v8f c = {};
#pragma unroll
    for (int kk = 0; kk < 16; ++kk) {
        const int cur = kk & 1;
        const int nxt = cur ^ 1;
        if (kk < 15) {
            const int i1 = 4 * (kk + 1) + half;
            a[nxt].q[0] = Hq[i1]; a[nxt].q[1] = Hq[i1 + 2];
            b[nxt].q[0] = Wq[i1]; b[nxt].q[1] = Wq[i1 + 2];
        }
        c = __builtin_amdgcn_wmma_f32_16x16x32_bf16(
                false, a[cur].v, false, b[cur].v, (short)0, c, false, false);
    }

    const float bi  = bhh[n0 + mn];
    const long  xstride = (long)SEQ * HID;   // row stride inside [B,T,H]
#pragma unroll
    for (int r = 0; r < 8; ++r) {
        const int  brow = m0 + r + 8 * half;
        float*     xp   = xbase + (long)brow * xstride + (n0 + mn);
        const float h   = tanhf(c[r] + *xp + bi);
        *xp = h;                                                   // outputs[b,t,:]
        hnext_b[(long)brow * HID + (n0 + mn)] = (__bf16)h;         // next-step A
        if (writeFinal)
            hfinal[(long)brow * HID + (n0 + mn)] = h;              // h_final
    }
}

// ---------------------------------------------------------------------------
// Host-side launch
//   inputs: 0=input(int) 1=hidden(f32) 2=emb(f32) 3=W_ih 4=W_hh 5=b_ih 6=b_hh
//   d_out : outputs [B,T,H] fp32  ++  h_final [B,H] fp32
//   d_ws  : | Wih_b | Whh_b | h_buf0 | h_buf1 |  (bf16, ~1.5 MB)
// ---------------------------------------------------------------------------
extern "C" void kernel_launch(void* const* d_in, const int* in_sizes, int n_in,
                              void* d_out, int out_size, void* d_ws, size_t ws_size,
                              hipStream_t stream) {
    const int*   input  = (const int*)d_in[0];
    const float* hidden = (const float*)d_in[1];
    const float* emb    = (const float*)d_in[2];
    const float* Wih    = (const float*)d_in[3];
    const float* Whh    = (const float*)d_in[4];
    const float* bih    = (const float*)d_in[5];
    const float* bhh    = (const float*)d_in[6];
    float* out = (float*)d_out;

    __bf16* Wih_b = (__bf16*)d_ws;
    __bf16* Whh_b = Wih_b + HID * HID;
    __bf16* h0b   = Whh_b + HID * HID;
    __bf16* h1b   = h0b + BATCH * HID;

    // Phase 0: weight + initial-hidden conversion to bf16
    prep_kernel<<<(HID * HID + 255) / 256, 256, 0, stream>>>(
        Wih, Whh, hidden, Wih_b, Whh_b, h0b);

    // Phase 1: fused embedding gather + input projection GEMM (WMMA bf16)
    xproj_kernel<<<dim3(MROWS / 16), 256, 0, stream>>>(
        input, emb, Wih_b, bih, out);

    // Phase 2: sequential recurrence, one launch per timestep
    float* hfinal = out + (long)MROWS * HID;
    __bf16* hcur = h0b;
    __bf16* hnxt = h1b;
    for (int t = 0; t < SEQ; ++t) {
        rnn_step_kernel<<<dim3(BATCH / 16, 4), 256, 0, stream>>>(
            hcur, Whh_b, bhh, out + (long)t * HID, hnxt, hfinal,
            (t == SEQ - 1) ? 1 : 0);
        __bf16* tmp = hcur; hcur = hnxt; hnxt = tmp;
    }
}
